// VRNN_32753420599586
// MI455X (gfx1250) — matrix operand order repaired and checked by reading, not compiled, not run on hardware
//
#include <hip/hip_runtime.h>
#include <cstdint>
#include <cstddef>

// ---------------- problem constants ----------------
#define T_STEPS 128
#define BATCH   256
#define XD      128
#define ZD      64
#define HD      512

// ---------------- vector types for WMMA ----------------
typedef __attribute__((ext_vector_type(16))) __bf16 v16bf;
typedef __attribute__((ext_vector_type(8)))  __bf16 v8bf;
typedef __attribute__((ext_vector_type(8)))  float  v8f;

// One K-segment of a segmented GEMM:  out[M,N] += A_seg[M,kw] @ W_seg[N,kw]^T
// A row-major, ld == kw; W row-major, leading dim ldw (pointer may be
// pre-offset into a wider [N, Ktot] matrix). kw == 0 -> segment unused.
struct Seg { const __bf16* A; const __bf16* W; int ldw; int kw; };

// One GEMM job for the multi-job kernel (selected by blockIdx.y).
struct Job { Seg s0; Seg s1; const float* bias; void* out; int N; };

__device__ __forceinline__ v16bf make16(v8bf lo, v8bf hi){
  v16bf r;
#pragma unroll
  for (int i=0;i<8;i++){ r[i]=lo[i]; r[i+8]=hi[i]; }
  return r;
}

// A fragment (16x32 bf16): lane lr holds M row; two 8-elem chunks:
//   K = k0 + lh*8 + [0..7]  and  K = k0 + 16 + lh*8 + [0..7]
__device__ __forceinline__ v16bf load_a(const Seg& s, int row, int k0, int lh){
  const __bf16* p = s.A + (size_t)row * s.kw + k0;
  return make16(*(const v8bf*)(p + lh*8), *(const v8bf*)(p + 16 + lh*8));
}
// B fragment (32x16 bf16): lane lr holds N col; K = k0 + lh*8 + [0..7] and k0 + 16 + lh*8 + [0..7]
__device__ __forceinline__ v16bf load_b(const Seg& s, int n, int k0, int lh){
  const __bf16* p = s.W + (size_t)n * s.ldw + k0;
  return make16(*(const v8bf*)(p + lh*8), *(const v8bf*)(p + 16 + lh*8));
}

#define WMMA_BF16(Va, Vb, Vc) \
  __builtin_amdgcn_wmma_f32_16x16x32_bf16(false, (Va), false, (Vb), (short)0, (Vc), false, false)

// ---------------------------------------------------------------------------
// Multi-job bf16 WMMA GEMM: out[256,N] = epi( sum_segs A@W^T + bias )
// blockIdx.y selects one of up to 4 independent jobs.
// Block = 256 threads = 8 waves; wave computes a 16(M) x 64(N) tile with
// V_WMMA_F32_16X16X32_BF16 (f32 accum); k-loop is software double-buffered
// so next iteration's 10 b128 loads are in flight during the 4 WMMAs.
// EPI: 0 = f32 linear store, 1 = bf16 ReLU store.
// Per-job blocks needed: (16 * N/64) / 8 = N/32 (grid.x = max over jobs).
// ---------------------------------------------------------------------------
template<int EPI>
__global__ __launch_bounds__(256)
void multi_gemm_kernel(Job j0, Job j1, Job j2, Job j3)
{
  Job jb = j0;
  if (blockIdx.y == 1) jb = j1;
  if (blockIdx.y == 2) jb = j2;
  if (blockIdx.y == 3) jb = j3;

  const int wave = threadIdx.x >> 5;
  const int lane = threadIdx.x & 31;
  const int lr   = lane & 15;       // row-in-tile (A: M, B: N)
  const int lh   = lane >> 4;       // half-wave selector (K phase)
  const int N    = jb.N;
  const int ntn  = N >> 6;          // # of 64-wide N tiles
  const int tile = blockIdx.x * 8 + wave;
  if (tile >= 16 * ntn) return;     // wave-uniform exit (EXEC stays all-1s)
  const int mt   = tile / ntn;
  const int nt   = tile - mt * ntn;
  const int m0   = mt << 4;
  const int n0   = nt << 6;

  v8f c[4];
#pragma unroll
  for (int j=0;j<4;j++)
#pragma unroll
    for (int i=0;i<8;i++) c[j][i] = 0.0f;

#pragma unroll
  for (int sg=0; sg<2; ++sg){
    const Seg s = sg ? jb.s1 : jb.s0;
    if (s.kw == 0) continue;

    // ---- software-pipelined k loop: prefetch (k0+32) during WMMAs of k0 ----
    v16bf a  = load_a(s, m0 + lr, 0, lh);
    v16bf b0 = load_b(s, n0 +  0 + lr, 0, lh);
    v16bf b1 = load_b(s, n0 + 16 + lr, 0, lh);
    v16bf b2 = load_b(s, n0 + 32 + lr, 0, lh);
    v16bf b3 = load_b(s, n0 + 48 + lr, 0, lh);
    for (int k0 = 32; k0 < s.kw; k0 += 32){
      v16bf an  = load_a(s, m0 + lr, k0, lh);
      v16bf bn0 = load_b(s, n0 +  0 + lr, k0, lh);
      v16bf bn1 = load_b(s, n0 + 16 + lr, k0, lh);
      v16bf bn2 = load_b(s, n0 + 32 + lr, k0, lh);
      v16bf bn3 = load_b(s, n0 + 48 + lr, k0, lh);
      c[0] = WMMA_BF16(a, b0, c[0]);
      c[1] = WMMA_BF16(a, b1, c[1]);
      c[2] = WMMA_BF16(a, b2, c[2]);
      c[3] = WMMA_BF16(a, b3, c[3]);
      a = an; b0 = bn0; b1 = bn1; b2 = bn2; b3 = bn3;
    }
    c[0] = WMMA_BF16(a, b0, c[0]);
    c[1] = WMMA_BF16(a, b1, c[1]);
    c[2] = WMMA_BF16(a, b2, c[2]);
    c[3] = WMMA_BF16(a, b3, c[3]);
  }

  // C/D layout: VGPR v, lanes 0-15 -> (M=v, N=lane); lanes 16-31 -> (M=v+8)
#pragma unroll
  for (int j=0;j<4;j++){
    const int col = n0 + j*16 + lr;
    const float bv = jb.bias ? jb.bias[col] : 0.0f;
#pragma unroll
    for (int v=0; v<8; v++){
      const int row = m0 + v + (lh << 3);
      float val = c[j][v] + bv;
      if (EPI == 1){
        val = fmaxf(val, 0.0f);
        ((__bf16*)jb.out)[(size_t)row * N + col] = (__bf16)val;
      } else {
        ((float*)jb.out)[(size_t)row * N + col] = val;
      }
    }
  }
}

// ---------------- elementwise helpers ----------------
__device__ __forceinline__ float clip6(float v){ return fminf(fmaxf(v,-6.0f),6.0f); }
__device__ __forceinline__ float sigm(float v){ return 1.0f/(1.0f+__expf(-v)); }

__global__ __launch_bounds__(256)
void cvt_kernel(const float* __restrict__ in, __bf16* __restrict__ out, int n){
  int i = blockIdx.x*256 + threadIdx.x;
  if (i < n) out[i] = (__bf16)in[i];
}

__global__ __launch_bounds__(256)
void init_kernel(float* acc, __bf16* h0, __bf16* h1){
  int i = blockIdx.x*256 + threadIdx.x;
  if (i < 2) acc[i] = 0.0f;
  if (i < BATCH*HD){ h0[i] = (__bf16)0.0f; h1[i] = (__bf16)0.0f; }
}

// z = mu_q + eps*exp(0.5*lv_q); KL accumulate (mean over batch)
__global__ __launch_bounds__(256)
void latent_kernel(const float* __restrict__ muq, const float* __restrict__ lvq,
                   const float* __restrict__ mup, const float* __restrict__ lvp,
                   const float* __restrict__ eps_t, __bf16* __restrict__ z,
                   float* kld_acc)
{
  __shared__ float red[256];
  int i = blockIdx.x*256 + threadIdx.x;           // i < 256*64
  float mq = muq[i];
  float lq = clip6(lvq[i]);
  float mp = mup[i];
  float lp = clip6(lvp[i]);
  z[i] = (__bf16)(mq + eps_t[i]*__expf(0.5f*lq));
  float d  = mq - mp;
  red[threadIdx.x] = 0.5f*(lp - lq + (__expf(lq) + d*d)*__expf(-lp) - 1.0f);
  __syncthreads();
  for (int s=128; s>0; s>>=1){
    if (threadIdx.x < s) red[threadIdx.x] += red[threadIdx.x+s];
    __syncthreads();
  }
  if (threadIdx.x==0) atomicAdd(kld_acc, red[0]*(1.0f/(float)BATCH));
}

// NLL accumulate (mean over batch)
__global__ __launch_bounds__(256)
void nll_kernel(const float* __restrict__ dmu, const float* __restrict__ dlv,
                const float* __restrict__ x_t, float* rec_acc)
{
  __shared__ float red[256];
  int i = blockIdx.x*256 + threadIdx.x;           // i < 256*128
  float lv = clip6(dlv[i]);
  float d  = x_t[i] - dmu[i];
  red[threadIdx.x] = 0.5f*(lv + d*d*__expf(-lv));
  __syncthreads();
  for (int s=128; s>0; s>>=1){
    if (threadIdx.x < s) red[threadIdx.x] += red[threadIdx.x+s];
    __syncthreads();
  }
  if (threadIdx.x==0) atomicAdd(rec_acc, red[0]*(1.0f/(float)BATCH));
}

// GRU gate math: GI = inp@Wih^T, GH = h@Whh^T (both [B, 3H] f32)
__global__ __launch_bounds__(256)
void gru_kernel(const float* __restrict__ GI, const float* __restrict__ GH,
                const __bf16* __restrict__ hprev, __bf16* __restrict__ hnew)
{
  int i = blockIdx.x*256 + threadIdx.x;           // i < 256*512
  int b = i >> 9;
  int j = i & 511;
  const float* gi = GI + (size_t)b*(3*HD);
  const float* gh = GH + (size_t)b*(3*HD);
  float r  = sigm(gi[j]         + gh[j]);
  float zg = sigm(gi[HD + j]    + gh[HD + j]);
  float n  = tanhf(gi[2*HD + j] + r*gh[2*HD + j]);
  float h  = (float)hprev[i];
  hnew[i]  = (__bf16)((1.0f - zg)*n + zg*h);
}

__global__ void finalize_kernel(const float* acc, float* out){
  if (threadIdx.x == 0){
    float kld = acc[0], rec = acc[1];
    const float invT = 1.0f/(float)T_STEPS;
    out[0] = (kld + rec)*invT;   // BETA = 1.0
    out[1] = rec*invT;
    out[2] = kld*invT;
  }
}

// ---------------------------------------------------------------------------
extern "C" void kernel_launch(void* const* d_in, const int* in_sizes, int n_in,
                              void* d_out, int out_size, void* d_ws, size_t ws_size,
                              hipStream_t stream)
{
  (void)n_in; (void)out_size; (void)ws_size;
  const float* x_f   = (const float*)d_in[0];
  const float* eps_f = (const float*)d_in[1];

  const float* phi_x_b1   = (const float*)d_in[3];
  const float* phi_x_b2   = (const float*)d_in[5];
  const float* phi_z_b    = (const float*)d_in[7];
  const float* enc_b1     = (const float*)d_in[9];
  const float* enc_b2     = (const float*)d_in[11];
  const float* enc_mu_b   = (const float*)d_in[13];
  const float* enc_lv_b   = (const float*)d_in[15];
  const float* prior_b    = (const float*)d_in[17];
  const float* prior_mu_b = (const float*)d_in[19];
  const float* prior_lv_b = (const float*)d_in[21];
  const float* dec_b1     = (const float*)d_in[23];
  const float* dec_b2     = (const float*)d_in[25];
  const float* dec_mu_b   = (const float*)d_in[27];
  const float* dec_lv_b   = (const float*)d_in[29];

  // ---- workspace allocator (256B aligned) ----
  size_t cur = 0;
  auto alloc = [&](size_t bytes)->void* {
    cur = (cur + 255) & ~(size_t)255;
    void* p = (char*)d_ws + cur;
    cur += bytes;
    return p;
  };

  // ---- one-time fp32 -> bf16 conversion of weights + x (L2-resident) ----
  const int widx[18] = {2,4,6,8,10,12,14,16,18,20,22,24,26,28,30,31,32,33};
  __bf16* wbf[34] = {};
  for (int i=0;i<18;i++){
    int id = widx[i];
    size_t n = (size_t)in_sizes[id];
    wbf[id] = (__bf16*)alloc(n*2);
    hipLaunchKernelGGL(cvt_kernel, dim3((unsigned)((n+255)/256)), dim3(256), 0, stream,
                       (const float*)d_in[id], wbf[id], (int)n);
  }
  size_t nx = (size_t)in_sizes[0];                 // T*B*X
  __bf16* xbf = (__bf16*)alloc(nx*2);
  hipLaunchKernelGGL(cvt_kernel, dim3((unsigned)((nx+255)/256)), dim3(256), 0, stream,
                     x_f, xbf, (int)nx);

  // ---- activation buffers ----
  __bf16* hx1   = (__bf16*)alloc((size_t)BATCH*HD*2);
  __bf16* phix  = (__bf16*)alloc((size_t)BATCH*HD*2);
  __bf16* prih  = (__bf16*)alloc((size_t)BATCH*HD*2);
  __bf16* enc1  = (__bf16*)alloc((size_t)BATCH*HD*2);
  __bf16* ench  = (__bf16*)alloc((size_t)BATCH*HD*2);
  __bf16* phiz  = (__bf16*)alloc((size_t)BATCH*HD*2);
  __bf16* dec1  = (__bf16*)alloc((size_t)BATCH*HD*2);
  __bf16* dech  = (__bf16*)alloc((size_t)BATCH*HD*2);
  __bf16* zbuf  = (__bf16*)alloc((size_t)BATCH*ZD*2);
  __bf16* h0A   = (__bf16*)alloc((size_t)BATCH*HD*2);
  __bf16* h0B   = (__bf16*)alloc((size_t)BATCH*HD*2);
  __bf16* h1A   = (__bf16*)alloc((size_t)BATCH*HD*2);
  __bf16* h1B   = (__bf16*)alloc((size_t)BATCH*HD*2);

  float* muq    = (float*)alloc((size_t)BATCH*ZD*4);
  float* lvq    = (float*)alloc((size_t)BATCH*ZD*4);
  float* mup    = (float*)alloc((size_t)BATCH*ZD*4);
  float* lvp    = (float*)alloc((size_t)BATCH*ZD*4);
  float* dmu    = (float*)alloc((size_t)BATCH*XD*4);
  float* dlv    = (float*)alloc((size_t)BATCH*XD*4);
  float* GI     = (float*)alloc((size_t)BATCH*3*HD*4);
  float* GH     = (float*)alloc((size_t)BATCH*3*HD*4);
  float* acc    = (float*)alloc(2*4);              // [kld, rec]

  hipLaunchKernelGGL(init_kernel, dim3(512), dim3(256), 0, stream, acc, h0A, h1A);

  const Seg snull{nullptr,nullptr,0,0};
  const Job jnull{snull, snull, nullptr, nullptr, 64};

  auto mkjob = [&](Seg s0, Seg s1, const float* bias, void* out, int N)->Job{
    return Job{s0, s1, bias, out, N};
  };
  // Launch 1..4 jobs (all same EPI); grid.x = max over jobs of N/32.
  auto gemm = [&](int epi, Job a, Job b = Job{ {nullptr,nullptr,0,0},{nullptr,nullptr,0,0},nullptr,nullptr,0 },
                  Job c = Job{ {nullptr,nullptr,0,0},{nullptr,nullptr,0,0},nullptr,nullptr,0 },
                  Job d = Job{ {nullptr,nullptr,0,0},{nullptr,nullptr,0,0},nullptr,nullptr,0 }){
    int nj = 1 + (b.out?1:0) + (c.out?1:0) + (d.out?1:0);
    int gx = a.N >> 5;
    if (b.out && (b.N>>5) > gx) gx = b.N>>5;
    if (c.out && (c.N>>5) > gx) gx = c.N>>5;
    if (d.out && (d.N>>5) > gx) gx = d.N>>5;
    if (!b.out) b = jnull; if (!c.out) c = jnull; if (!d.out) d = jnull;
    dim3 g((unsigned)gx, (unsigned)nj), blk(256);
    if (epi == 1) hipLaunchKernelGGL((multi_gemm_kernel<1>), g, blk, 0, stream, a, b, c, d);
    else          hipLaunchKernelGGL((multi_gemm_kernel<0>), g, blk, 0, stream, a, b, c, d);
  };

  __bf16 *h0c = h0A, *h0n = h0B, *h1c = h1A, *h1n = h1B;

  for (int t=0; t<T_STEPS; ++t){
    const __bf16* x_t   = xbf   + (size_t)t*BATCH*XD;
    const float*  xf_t  = x_f   + (size_t)t*BATCH*XD;
    const float*  eps_t = eps_f + (size_t)t*BATCH*ZD;

    // [phi_x layer1, prior trunk] -- independent, one launch
    gemm(1, mkjob(Seg{x_t, wbf[2], XD, XD}, snull, phi_x_b1, hx1, HD),
            mkjob(Seg{h1c, wbf[16], HD, HD}, snull, prior_b,  prih, HD));
    // phi_x layer2
    gemm(1, mkjob(Seg{hx1, wbf[4], HD, HD}, snull, phi_x_b2, phix, HD));
    // encoder MLP on [phi_x | h_last]
    gemm(1, mkjob(Seg{phix, wbf[8],      2*HD, HD},
                  Seg{h1c,  wbf[8] + HD, 2*HD, HD}, enc_b1, enc1, HD));
    gemm(1, mkjob(Seg{enc1, wbf[10], HD, HD}, snull, enc_b2, ench, HD));
    // 4 latent heads (f32), one launch
    gemm(0, mkjob(Seg{ench, wbf[12], HD, HD}, snull, enc_mu_b,   muq, ZD),
            mkjob(Seg{ench, wbf[14], HD, HD}, snull, enc_lv_b,   lvq, ZD),
            mkjob(Seg{prih, wbf[18], HD, HD}, snull, prior_mu_b, mup, ZD),
            mkjob(Seg{prih, wbf[20], HD, HD}, snull, prior_lv_b, lvp, ZD));
    // reparameterize + KL
    hipLaunchKernelGGL(latent_kernel, dim3(BATCH*ZD/256), dim3(256), 0, stream,
                       muq, lvq, mup, lvp, eps_t, zbuf, acc + 0);
    // phi_z
    gemm(1, mkjob(Seg{zbuf, wbf[6], ZD, ZD}, snull, phi_z_b, phiz, HD));
    // decoder MLP on [phi_z | h_last]
    gemm(1, mkjob(Seg{phiz, wbf[22],      2*HD, HD},
                  Seg{h1c,  wbf[22] + HD, 2*HD, HD}, dec_b1, dec1, HD));
    gemm(1, mkjob(Seg{dec1, wbf[24], HD, HD}, snull, dec_b2, dech, HD));
    // 2 decoder heads (f32), one launch; then NLL
    gemm(0, mkjob(Seg{dech, wbf[26], HD, HD}, snull, dec_mu_b, dmu, XD),
            mkjob(Seg{dech, wbf[28], HD, HD}, snull, dec_lv_b, dlv, XD));
    hipLaunchKernelGGL(nll_kernel, dim3(BATCH*XD/256), dim3(256), 0, stream,
                       dmu, dlv, xf_t, acc + 1);
    // GRU layer 0:  GI = [phi_x | phi_z] @ Wih0^T,  GH = h0 @ Whh0^T
    gemm(0, mkjob(Seg{phix, wbf[30],      2*HD, HD},
                  Seg{phiz, wbf[30] + HD, 2*HD, HD}, nullptr, GI, 3*HD),
            mkjob(Seg{h0c,  wbf[31], HD, HD}, snull, nullptr, GH, 3*HD));
    hipLaunchKernelGGL(gru_kernel, dim3(BATCH*HD/256), dim3(256), 0, stream,
                       GI, GH, h0c, h0n);
    // GRU layer 1:  input = new h0
    gemm(0, mkjob(Seg{h0n, wbf[32], HD, HD}, snull, nullptr, GI, 3*HD),
            mkjob(Seg{h1c, wbf[33], HD, HD}, snull, nullptr, GH, 3*HD));
    hipLaunchKernelGGL(gru_kernel, dim3(BATCH*HD/256), dim3(256), 0, stream,
                       GI, GH, h1c, h1n);

    { __bf16* tmp = h0c; h0c = h0n; h0n = tmp; }
    { __bf16* tmp = h1c; h1c = h1n; h1n = tmp; }
  }

  hipLaunchKernelGGL(finalize_kernel, dim3(1), dim3(64), 0, stream,
                     acc, (float*)d_out);
}
